// MoELoRAqkv_3805341024605
// MI455X (gfx1250) — matrix operand order repaired and checked by reading, not compiled, not run on hardware
//
#include <hip/hip_runtime.h>

// ---------------------------------------------------------------------------
// MoE-LoRA fused QKV for MI455X (gfx1250, wave32, WMMA).
// Main GEMM: bf16 operands, f32 accumulate via v_wmma_f32_16x16x32_bf16.
// LoRA (rank16 x k2) folded in as one extra K=32 WMMA per output tile.
// Double-buffered LDS (1 barrier/K-step); v_perm_b32 packed bf16 conversion.
// ---------------------------------------------------------------------------

typedef __attribute__((ext_vector_type(16))) __bf16 v16bf;
typedef __attribute__((ext_vector_type(8)))  __bf16 v8bf;
typedef __attribute__((ext_vector_type(8)))  float  v8f;
typedef __attribute__((ext_vector_type(4)))  float  f32x4;

#define DIMK   1024
#define NDIM   3072
#define MDIM   16384   // 32 * 512
#define BM     128
#define BN     128
#define BK     32
#define LDT    40      // padded LDS row pitch in bf16 elems (80B: 16B aligned, conflict-free)

// Pack two f32 -> two bf16 in one dword: bias-add (round-half-up) + one v_perm_b32.
__device__ __forceinline__ unsigned pk_bf16(float a, float b) {
    unsigned ua = __builtin_bit_cast(unsigned, a) + 0x8000u;
    unsigned ub = __builtin_bit_cast(unsigned, b) + 0x8000u;
    // result = { ub[31:16], ua[31:16] } : sel bytes {2,3} of S1=ua, {2,3} of S0=ub
    return __builtin_amdgcn_perm(ub, ua, 0x07060302u);
}

__device__ __forceinline__ v16bf cat8(v8bf lo, v8bf hi) {
    return __builtin_shufflevector(lo, hi, 0,1,2,3,4,5,6,7,8,9,10,11,12,13,14,15);
}

// ---------------------------------------------------------------------------
// Kernel 1: T[sel][m][k*16+r] = sum_i x[m,i] * A_pool[sel][idx[b,k], i, r]  (bf16 out)
// One thread per (sel,k,m); lanes vary over m -> wave-uniform A loads.
// ---------------------------------------------------------------------------
__global__ __launch_bounds__(256)
void moe_lora_t(const float* __restrict__ x, const float* __restrict__ Aq,
                const float* __restrict__ Av, const int* __restrict__ idx,
                unsigned short* __restrict__ Tq, unsigned short* __restrict__ Tv)
{
    const int gid = blockIdx.x * 256 + threadIdx.x;   // 65536 total
    const int m   = gid & (MDIM - 1);
    const int t   = gid >> 14;
    const int k   = t & 1;
    const int sel = t >> 1;
    const int b   = m >> 9;                           // 512 rows per batch
    const int e   = idx[b * 2 + k];

    const float* A  = (sel ? Av : Aq) + (size_t)e * (DIMK * 16);
    const float* xr = x + (size_t)m * DIMK;

    float acc[16];
#pragma unroll
    for (int r = 0; r < 16; ++r) acc[r] = 0.f;

    for (int i = 0; i < DIMK; ++i) {
        const float xv = xr[i];
        const f32x4* Ar = (const f32x4*)(A + (size_t)i * 16);
#pragma unroll
        for (int q4 = 0; q4 < 4; ++q4) {
            f32x4 v = Ar[q4];
            acc[q4 * 4 + 0] += xv * v[0];
            acc[q4 * 4 + 1] += xv * v[1];
            acc[q4 * 4 + 2] += xv * v[2];
            acc[q4 * 4 + 3] += xv * v[3];
        }
    }

    unsigned short* T = (sel ? Tv : Tq) + (size_t)m * 32 + k * 16;
#pragma unroll
    for (int r = 0; r < 16; r += 2)
        *(unsigned*)(T + r) = pk_bf16(acc[r], acc[r + 1]);
}

// ---------------------------------------------------------------------------
// Kernel 2: Bct[sel][b][o][k*16+r] = attn[b,k] * B_pool[sel][idx[b,k], r, o]  (bf16)
// Transposed (o-major, contiguous kr) to match the WMMA B-fragment layout.
// attn == 0 masking is implicit in the multiply.
// ---------------------------------------------------------------------------
__global__ __launch_bounds__(256)
void moe_lora_bct(const float* __restrict__ Bq, const float* __restrict__ Bv,
                  const float* __restrict__ attn, const int* __restrict__ idx,
                  unsigned short* __restrict__ Bctq, unsigned short* __restrict__ Bctv)
{
    const int gid = blockIdx.x * 256 + threadIdx.x;   // 65536 total
    const int o   = gid & 1023;
    const int t   = gid >> 10;
    const int b   = t & 31;
    const int sel = t >> 5;

    const float* B = sel ? Bv : Bq;
    unsigned short* Bct = (sel ? Bctv : Bctq) + ((size_t)b * 1024 + o) * 32;

#pragma unroll
    for (int k = 0; k < 2; ++k) {
        const int   e = idx[b * 2 + k];
        const float a = attn[b * 2 + k];
        const float* Brow = B + (size_t)e * 16 * 1024 + o;
#pragma unroll
        for (int r = 0; r < 16; r += 2) {
            float v0 = a * Brow[(size_t)r * 1024];
            float v1 = a * Brow[(size_t)(r + 1) * 1024];
            *(unsigned*)(Bct + k * 16 + r) = pk_bf16(v0, v1);
        }
    }
}

// ---------------------------------------------------------------------------
// Kernel 3: cb[b][o] = sum_k attn[b,k] * bias_pool[idx[b,k], o]
// ---------------------------------------------------------------------------
__global__ __launch_bounds__(256)
void moe_cb(const float* __restrict__ bias_pool, const float* __restrict__ attn,
            const int* __restrict__ idx, float* __restrict__ cb)
{
    const int gid = blockIdx.x * 256 + threadIdx.x;   // 98304 total
    const int o = gid % NDIM;
    const int b = gid / NDIM;
    cb[gid] = attn[b * 2 + 0] * bias_pool[(size_t)idx[b * 2 + 0] * NDIM + o]
            + attn[b * 2 + 1] * bias_pool[(size_t)idx[b * 2 + 1] * NDIM + o];
}

// ---------------------------------------------------------------------------
// Kernel 4: main GEMM + LoRA WMMA epilogue + bias epilogue.
// Block: 256 threads = 8 waves (4 in M x 2 in N); block tile 128x128, BK=32.
// Wave tile 32(M) x 64(N) => 2x4 accumulators of v8f. Double-buffered LDS.
// ---------------------------------------------------------------------------
__global__ __launch_bounds__(256, 2)
void moe_qkv_main(const float* __restrict__ x, const float* __restrict__ w,
                  const float* __restrict__ bias, const float* __restrict__ cb,
                  const unsigned short* __restrict__ Tq, const unsigned short* __restrict__ Tv,
                  const unsigned short* __restrict__ Bctq, const unsigned short* __restrict__ Bctv,
                  float* __restrict__ out)
{
    __shared__ unsigned short Alds[2][BM * LDT];   // 2 x 10 KB (bf16 as ushort)
    __shared__ unsigned short Blds[2][BN * LDT];   // 2 x 10 KB

    const int tid  = threadIdx.x;
    const int nblk = blockIdx.x * BN;
    const int mblk = blockIdx.y * BM;
    const int b    = mblk >> 9;                 // batch index (512 | BM spacing)

    const int lane = tid & 31;
    const int wave = tid >> 5;
    const int half = lane >> 4;                 // 0/1 half-wave (WMMA layout)
    const int lm   = lane & 15;
    const int wm   = (wave & 3) * 32;           // wave M offset in block tile
    const int wn   = (wave >> 2) * 64;          // wave N offset in block tile

    // per-thread global->LDS mapping: 4 float4 per tile (128 rows x 8 float4)
    int rrow[4], rc4[4];
#pragma unroll
    for (int i = 0; i < 4; ++i) { int li = tid + i * 256; rrow[i] = li >> 3; rc4[i] = li & 7; }

    f32x4 ra[4], rb[4];

    auto gload = [&](int kk) {
#pragma unroll
        for (int i = 0; i < 4; ++i) {
            ra[i] = *(const f32x4*)(x + (size_t)(mblk + rrow[i]) * DIMK + kk + rc4[i] * 4);
            rb[i] = *(const f32x4*)(w + (size_t)(nblk + rrow[i]) * DIMK + kk + rc4[i] * 4);
        }
        if (kk + 64 < DIMK) {   // hint the K-block after next into L2/WGP$
            __builtin_prefetch(x + (size_t)(mblk + rrow[0]) * DIMK + kk + 64, 0, 0);
            __builtin_prefetch(w + (size_t)(nblk + rrow[0]) * DIMK + kk + 64, 0, 0);
        }
    };

    auto s2lds = [&](int buf) {
#pragma unroll
        for (int i = 0; i < 4; ++i) {
            uint2 pa, pb;
            pa.x = pk_bf16(ra[i][0], ra[i][1]);
            pa.y = pk_bf16(ra[i][2], ra[i][3]);
            *(uint2*)(&Alds[buf][rrow[i] * LDT + rc4[i] * 4]) = pa;
            pb.x = pk_bf16(rb[i][0], rb[i][1]);
            pb.y = pk_bf16(rb[i][2], rb[i][3]);
            *(uint2*)(&Blds[buf][rrow[i] * LDT + rc4[i] * 4]) = pb;
        }
    };

    v8f acc[2][4];
#pragma unroll
    for (int i = 0; i < 2; ++i)
#pragma unroll
        for (int j = 0; j < 4; ++j)
#pragma unroll
            for (int e = 0; e < 8; ++e) acc[i][j][e] = 0.f;

    gload(0);
    s2lds(0);
    __syncthreads();

    int buf = 0;
    for (int kk = 0; kk < DIMK; kk += BK) {
        const bool more = (kk + BK < DIMK);
        if (more) gload(kk + BK);               // global loads overlap this tile's WMMA

        v16bf af[2], bfr[4];
#pragma unroll
        for (int ms = 0; ms < 2; ++ms) {
            // 16-bit A 16x32: elems 0-7 -> K=half*8+0..7, elems 8-15 -> K=half*8+16..23
            const unsigned short* p = &Alds[buf][(wm + ms * 16 + lm) * LDT + half * 8];
            af[ms] = cat8(*(const v8bf*)(const void*)p, *(const v8bf*)(const void*)(p + 16));
        }
#pragma unroll
        for (int ns = 0; ns < 4; ++ns) {
            // 16-bit B 32x16: elems 0-7 -> K=half*8+0..7, elems 8-15 -> K=half*8+16..23
            const unsigned short* p = &Blds[buf][(wn + ns * 16 + lm) * LDT + half * 8];
            bfr[ns] = cat8(*(const v8bf*)(const void*)p, *(const v8bf*)(const void*)(p + 16));
        }
#pragma unroll
        for (int ms = 0; ms < 2; ++ms)
#pragma unroll
            for (int ns = 0; ns < 4; ++ns)
                acc[ms][ns] = __builtin_amdgcn_wmma_f32_16x16x32_bf16(
                    false, af[ms], false, bfr[ns], (short)0, acc[ms][ns], false, false);

        if (more) s2lds(buf ^ 1);               // fill other buffer while WMMAs drain
        __syncthreads();
        buf ^= 1;
    }

    // ---- LoRA epilogue: one K=32 WMMA per tile (q third and v third only) ----
    const int third = nblk >> 10;               // 0=q, 1=k(skip), 2=v; block-uniform
    if (third != 1) {
        const unsigned short* T  = (third == 0) ? Tq : Tv;
        const unsigned short* Bc = (third == 0) ? Bctq : Bctv;
        const int ob = (third == 2) ? (nblk - 2048) : nblk;

        v16bf atf[2];
#pragma unroll
        for (int ms = 0; ms < 2; ++ms) {
            const unsigned short* p = T + (size_t)(mblk + wm + ms * 16 + lm) * 32 + half * 8;
            atf[ms] = cat8(*(const v8bf*)(const void*)p, *(const v8bf*)(const void*)(p + 16));
        }
#pragma unroll
        for (int ns = 0; ns < 4; ++ns) {
            const unsigned short* p =
                Bc + ((size_t)b * 1024 + (ob + wn + ns * 16 + lm)) * 32 + half * 8;
            v16bf bff = cat8(*(const v8bf*)(const void*)p, *(const v8bf*)(const void*)(p + 16));
#pragma unroll
            for (int ms = 0; ms < 2; ++ms)
                acc[ms][ns] = __builtin_amdgcn_wmma_f32_16x16x32_bf16(
                    false, atf[ms], false, bff, (short)0, acc[ms][ns], false, false);
        }
    }

    // ---- store: C layout VGPR r, lane l -> M = r + 8*half, N = lm ----
#pragma unroll
    for (int ms = 0; ms < 2; ++ms)
#pragma unroll
        for (int ns = 0; ns < 4; ++ns) {
            const int gm = mblk + wm + ms * 16 + half * 8;
            const int gn = nblk + wn + ns * 16 + lm;
            const float badd = bias[gn] + cb[b * NDIM + gn];
            float* po = out + (size_t)gm * NDIM + gn;
#pragma unroll
            for (int r = 0; r < 8; ++r)
                po[(size_t)r * NDIM] = acc[ms][ns][r] + badd;
        }
}

// ---------------------------------------------------------------------------
// Host launcher. ws layout (bytes):
//   [0,1M)  Tq   [1M,2M) Tv   [2M,4M) Bctq   [4M,6M) Bctv   [6M,6M+384K) cb
// ---------------------------------------------------------------------------
extern "C" void kernel_launch(void* const* d_in, const int* in_sizes, int n_in,
                              void* d_out, int out_size, void* d_ws, size_t ws_size,
                              hipStream_t stream)
{
    (void)in_sizes; (void)n_in; (void)out_size; (void)ws_size;

    const float* x      = (const float*)d_in[0];
    const float* weight = (const float*)d_in[1];
    const float* bias   = (const float*)d_in[2];
    const float* Aq     = (const float*)d_in[3];
    const float* Bq     = (const float*)d_in[4];
    const float* Av     = (const float*)d_in[5];
    const float* Bv     = (const float*)d_in[6];
    const float* bpool  = (const float*)d_in[7];
    const float* attn   = (const float*)d_in[8];
    const int*   idx    = (const int*)d_in[9];
    float* out = (float*)d_out;

    char* ws = (char*)d_ws;
    unsigned short* Tq   = (unsigned short*)(ws + (0u << 20));
    unsigned short* Tv   = (unsigned short*)(ws + (1u << 20));
    unsigned short* Bctq = (unsigned short*)(ws + (2u << 20));
    unsigned short* Bctv = (unsigned short*)(ws + (4u << 20));
    float*          cb   = (float*)(ws + (6u << 20));

    moe_lora_t  <<<65536 / 256, 256, 0, stream>>>(x, Aq, Av, idx, Tq, Tv);
    moe_lora_bct<<<65536 / 256, 256, 0, stream>>>(Bq, Bv, attn, idx, Bctq, Bctv);
    moe_cb      <<<98304 / 256, 256, 0, stream>>>(bpool, attn, idx, cb);

    dim3 grid(NDIM / BN, MDIM / BM);   // 24 x 128 = 3072 workgroups
    moe_qkv_main<<<grid, 256, 0, stream>>>(x, weight, bias, cb, Tq, Tv, Bctq, Bctv, out);
}
